// SwinIR_128849019416
// MI455X (gfx1250) — compile-verified, matrix operand order checked by reading, not run on hardware
//
#include <hip/hip_runtime.h>
#include <hip/hip_bf16.h>
#include <math.h>

typedef __attribute__((ext_vector_type(16))) _Float16 v16h;
typedef __attribute__((ext_vector_type(8)))  _Float16 v8h;
typedef __attribute__((ext_vector_type(8)))  float    v8f;

#define DIM    180
#define NQKV   540
#define KP     192            // DIM padded to multiple of 32 (WMMA K)
#define HEADS  6
#define HD     30
#define HDP    32             // head dim padded to 32
#define NTOK   64             // tokens per window
#define NWIN   4096           // B * NW
#define NMASK  1024           // windows per image
#define MROWS  (NWIN * NTOK)  // 262144

// ---------------------------------------------------------------------------
// CDNA5 async global->LDS copy (ASYNCcnt path); guarded for portability.
// Builtin pointer params are 16-byte int-vector typed (per compiler probe).
// ---------------------------------------------------------------------------
#if defined(__has_builtin)
#if __has_builtin(__builtin_amdgcn_global_load_async_to_lds_b128)
#define HAVE_ASYNC_LDS 1
#endif
#endif

#if defined(HAVE_ASYNC_LDS)
typedef int v4i_async __attribute__((vector_size(4 * sizeof(int))));
__device__ __forceinline__ void async_copy16(const void* gsrc, void* ldst) {
  __builtin_amdgcn_global_load_async_to_lds_b128(
      (__attribute__((address_space(1))) v4i_async*)gsrc,
      (__attribute__((address_space(3))) v4i_async*)ldst, 0, 0);
}
__device__ __forceinline__ void wait_async0() {
#if __has_builtin(__builtin_amdgcn_s_wait_asynccnt)
  __builtin_amdgcn_s_wait_asynccnt(0);
#else
  asm volatile("s_wait_asynccnt 0x0" ::: "memory");
#endif
}
#endif

// ---------------------------------------------------------------------------
// Fragment loader.  Per CDNA5 ISA (CDNA5 ISA, 16-bit A 16x32 layout):
// lanes 0-15 hold K=[0..7] / K=[16..23], lanes 16-31 hold K=[8..15] /
// K=[24..31]: two contiguous 8-half chunks per lane.  B uses the
// transpose-symmetric layout from an [N][K] row-major buffer.
// ---------------------------------------------------------------------------
__device__ __forceinline__ v16h load_frag_h(const _Float16* src, int stride, int lane) {
  int r = lane & 15;
  int hi = lane >> 4;
  const _Float16* p = src + (size_t)r * stride + hi * 8;
  v8h lo = *(const v8h*)(p);
  v8h hh = *(const v8h*)(p + 16);
  v16h o;
#pragma unroll
  for (int j = 0; j < 8; ++j) { o[j] = lo[j]; o[j + 8] = hh[j]; }
  return o;
}

// ---------------------------------------------------------------------------
// Kernel 1: weight transpose/pad to f16 + relative-position bias gather.
// ---------------------------------------------------------------------------
__global__ void prep_kernel(const float* __restrict__ qkv_w, const float* __restrict__ proj_w,
                            const float* __restrict__ bias_table, const int* __restrict__ rel_index,
                            _Float16* __restrict__ qkv_wt, _Float16* __restrict__ proj_wt,
                            float* __restrict__ rel_bias) {
  int i = blockIdx.x * blockDim.x + threadIdx.x;
  if (i < NQKV * KP) {                       // qkv_wt[n][k] = qkv_w[k][n], K padded
    int n = i / KP, k = i % KP;
    qkv_wt[i] = (k < DIM) ? (_Float16)qkv_w[(size_t)k * NQKV + n] : (_Float16)0.0f;
  }
  if (i < DIM * KP) {                        // proj_wt[n][k] = proj_w[k][n]
    int n = i / KP, k = i % KP;
    proj_wt[i] = (k < DIM) ? (_Float16)proj_w[(size_t)k * DIM + n] : (_Float16)0.0f;
  }
  if (i < HEADS * NTOK * NTOK) {             // rel_bias[h][i][j]
    int h = i / (NTOK * NTOK), ij = i % (NTOK * NTOK);
    rel_bias[i] = bias_table[rel_index[ij] * HEADS + h];
  }
}

// ---------------------------------------------------------------------------
// Kernel 2: QKV GEMM  (262144 x 540 = X(262144x180) * W(180x540) + b)
// One block per 32 M-rows covering ALL 540 N-columns; x converted to f16
// once into LDS (read exactly once from HBM).  Each wave owns one fixed
// M-subtile (msub = wave&1, loop-invariant) -> its 6 A-fragments stay in
// VGPRs; B-fragments are software-pipelined one WMMA ahead.
// ---------------------------------------------------------------------------
__global__ void __launch_bounds__(256) qkv_gemm_kernel(
    const float* __restrict__ x, const _Float16* __restrict__ wt,
    const float* __restrict__ qkv_b,
    _Float16* __restrict__ qb, _Float16* __restrict__ kb, _Float16* __restrict__ vtb) {
  __shared__ _Float16 Ah[32][KP];
  int lane = threadIdx.x & 31;
  int wave = threadIdx.x >> 5;
  int m0 = blockIdx.x * 32;

  // stage + convert the 32x180 fp32 x tile into 32x192 f16 LDS (zero-padded)
  const float* xrow = x + (size_t)m0 * DIM;
  for (int idx = threadIdx.x; idx < 32 * (KP / 2); idx += 256) {
    int r = idx / (KP / 2);
    int k = (idx % (KP / 2)) * 2;
    float a0 = (k     < DIM) ? xrow[(size_t)r * DIM + k]     : 0.0f;
    float a1 = (k + 1 < DIM) ? xrow[(size_t)r * DIM + k + 1] : 0.0f;
    Ah[r][k]     = (_Float16)a0;
    Ah[r][k + 1] = (_Float16)a1;
  }
  __syncthreads();

  int msub = wave & 1;                       // invariant: t = wave + 8j  =>  t&1 == wave&1
  v16h afrag[6];
#pragma unroll
  for (int ks = 0; ks < 6; ++ks)
    afrag[ks] = load_frag_h(&Ah[msub * 16][ks * 32], KP, lane);

  const float scale = 0.18257418583505536f;  // 30^-0.5
  v16h bcur = load_frag_h(wt + (size_t)((wave >> 1) * 16) * KP, KP, lane);
  for (int t = wave; t < 68; t += 8) {
    int n0 = (t >> 1) * 16;
    const _Float16* wn = wt + (size_t)n0 * KP;
    int tn = (t + 8 < 68) ? (t + 8) : t;     // next tile (clamped for last iter)
    const _Float16* wn_next = wt + (size_t)((tn >> 1) * 16) * KP;

    v8f acc = {};
#pragma unroll
    for (int ks = 0; ks < 6; ++ks) {
      v16h bnext;
      if (ks < 5) bnext = load_frag_h(wn + (ks + 1) * 32, KP, lane);
      else        bnext = load_frag_h(wn_next, KP, lane);
      acc = __builtin_amdgcn_wmma_f32_16x16x32_f16(false, afrag[ks], false, bcur, (short)0, acc, false, false);
      bcur = bnext;
    }

    int col = n0 + (lane & 15);
    if (col < NQKV) {
      int which = col / DIM;
      int h = (col % DIM) / HD;
      int d = col % HD;
      float bias = qkv_b[col];
#pragma unroll
      for (int r = 0; r < 8; ++r) {
        int row = m0 + msub * 16 + r + 8 * (lane >> 4);
        int w = row >> 6;
        int tk = row & 63;
        float v = acc[r] + bias;
        size_t wh = (size_t)(w * HEADS + h);
        if (which == 0) {
          _Float16* p = qb + (wh * NTOK + tk) * HDP;
          p[d] = (_Float16)(v * scale);
          if (d == HD - 1) { p[HD] = (_Float16)0.0f; p[HD + 1] = (_Float16)0.0f; }
        } else if (which == 1) {
          _Float16* p = kb + (wh * NTOK + tk) * HDP;
          p[d] = (_Float16)v;
          if (d == HD - 1) { p[HD] = (_Float16)0.0f; p[HD + 1] = (_Float16)0.0f; }
        } else {
          vtb[(wh * HDP + d) * NTOK + tk] = (_Float16)v;
        }
      }
    }
  }
}

// ---------------------------------------------------------------------------
// Kernel 3: fused window attention.  One block (8 waves) per window, loops
// over 6 heads.  S and softmax(P) live entirely in LDS (24 KB); the 64x64
// logits never touch HBM.
// ---------------------------------------------------------------------------
__global__ void __launch_bounds__(256) attn_kernel(
    const _Float16* __restrict__ qb, const _Float16* __restrict__ kb,
    const _Float16* __restrict__ vtb, const float* __restrict__ rel_bias,
    const float* __restrict__ mask, _Float16* __restrict__ ao) {
  __shared__ float    S[NTOK][NTOK];
  __shared__ _Float16 P[NTOK][NTOK];
  int w = blockIdx.x;
  int lane = threadIdx.x & 31;
  int wave = threadIdx.x >> 5;

  // zero the K-padding columns [180,192) of this window's activation rows
  for (int i = threadIdx.x; i < NTOK * (KP - DIM); i += 256) {
    int t = i / (KP - DIM), c = DIM + i % (KP - DIM);
    ao[((size_t)w * NTOK + t) * KP + c] = (_Float16)0.0f;
  }

  const float* mrow = mask + (size_t)(w % NMASK) * NTOK * NTOK;

  for (int h = 0; h < HEADS; ++h) {
    const _Float16* qh = qb  + ((size_t)(w * HEADS + h)) * NTOK * HDP;
    const _Float16* kh = kb  + ((size_t)(w * HEADS + h)) * NTOK * HDP;
    const _Float16* vh = vtb + ((size_t)(w * HEADS + h)) * HDP * NTOK;
    const float*    bh = rel_bias + (size_t)h * NTOK * NTOK;

    // ---- S = q @ k^T : 16 tiles of 16x16, K=32 in one WMMA; 2 tiles/wave ----
#pragma unroll
    for (int tt = 0; tt < 2; ++tt) {
      int tile = wave * 2 + tt;
      int mi = (tile >> 2) * 16, ni = (tile & 3) * 16;
      v16h a = load_frag_h(qh + (size_t)mi * HDP, HDP, lane);
      v16h b = load_frag_h(kh + (size_t)ni * HDP, HDP, lane);
      v8f acc = {};
      acc = __builtin_amdgcn_wmma_f32_16x16x32_f16(false, a, false, b, (short)0, acc, false, false);
      int col = ni + (lane & 15);
#pragma unroll
      for (int r = 0; r < 8; ++r) {
        int row = mi + r + 8 * (lane >> 4);
        S[row][col] = acc[r] + bh[row * NTOK + col] + mrow[row * NTOK + col];
      }
    }
    __syncthreads();

    // ---- softmax: 4 lanes per row, width-4 shuffle reductions ----
    {
      int row = threadIdx.x >> 2;
      int c0 = (threadIdx.x & 3) * 16;
      float m = -3.0e38f;
#pragma unroll
      for (int j = 0; j < 16; ++j) m = fmaxf(m, S[row][c0 + j]);
      m = fmaxf(m, __shfl_xor(m, 1, 4));
      m = fmaxf(m, __shfl_xor(m, 2, 4));
      float e[16], s = 0.0f;
#pragma unroll
      for (int j = 0; j < 16; ++j) { e[j] = __expf(S[row][c0 + j] - m); s += e[j]; }
      s += __shfl_xor(s, 1, 4);
      s += __shfl_xor(s, 2, 4);
      float inv = 1.0f / s;
#pragma unroll
      for (int j = 0; j < 16; ++j) P[row][c0 + j] = (_Float16)(e[j] * inv);
    }
    __syncthreads();

    // ---- O = P @ V : 8 tiles (4M x 2N), K=64 in 2 WMMA steps; 1 tile/wave ----
    {
      int mi = (wave >> 1) * 16, ni = (wave & 1) * 16;
      v8f acc = {};
#pragma unroll
      for (int ks = 0; ks < 2; ++ks) {
        int k0 = ks * 32;
        v16h a = load_frag_h(&P[mi][k0], NTOK, lane);                  // from LDS
        v16h b = load_frag_h(vh + (size_t)ni * NTOK + k0, NTOK, lane); // V^T rows = dims
        acc = __builtin_amdgcn_wmma_f32_16x16x32_f16(false, a, false, b, (short)0, acc, false, false);
      }
      int d = ni + (lane & 15);
      if (d < HD) {
#pragma unroll
        for (int r = 0; r < 8; ++r) {
          int t = mi + r + 8 * (lane >> 4);
          ao[((size_t)w * NTOK + t) * KP + h * HD + d] = (_Float16)acc[r];
        }
      }
    }
    __syncthreads();
  }
}

// ---------------------------------------------------------------------------
// Kernel 4: output projection (262144 x 180 = AO(262144x192) * Wp(192x180)+b)
// A tile staged to LDS via GLOBAL_LOAD_ASYNC_TO_LDS_B128 (ASYNCcnt path);
// per-wave fixed M-subtile with resident A-fragments, pipelined B.
// ---------------------------------------------------------------------------
__global__ void __launch_bounds__(256) proj_gemm_kernel(
    const _Float16* __restrict__ ao, const _Float16* __restrict__ wt,
    const float* __restrict__ proj_b, float* __restrict__ out) {
  __shared__ _Float16 Ah[32][KP];
  int lane = threadIdx.x & 31;
  int wave = threadIdx.x >> 5;
  int m0 = blockIdx.x * 32;

  // stage the 32x192 f16 tile (rows are contiguous: 12288 B flat copy)
#if defined(HAVE_ASYNC_LDS)
  {
    const char* g = (const char*)(ao + (size_t)m0 * KP);
    char* l = (char*)&Ah[0][0];
    for (int idx = threadIdx.x; idx < (32 * KP * 2) / 16; idx += 256)
      async_copy16(g + idx * 16, l + idx * 16);
    wait_async0();
  }
#else
  {
    const v8h* g = (const v8h*)(ao + (size_t)m0 * KP);
    v8h* l = (v8h*)&Ah[0][0];
    for (int idx = threadIdx.x; idx < (32 * KP) / 8; idx += 256) l[idx] = g[idx];
  }
#endif
  __syncthreads();

  int msub = wave & 1;
  v16h afrag[6];
#pragma unroll
  for (int ks = 0; ks < 6; ++ks)
    afrag[ks] = load_frag_h(&Ah[msub * 16][ks * 32], KP, lane);

  v16h bcur = load_frag_h(wt + (size_t)((wave >> 1) * 16) * KP, KP, lane);
  for (int t = wave; t < 24; t += 8) {
    int n0 = (t >> 1) * 16;
    const _Float16* wn = wt + (size_t)n0 * KP;
    int tn = (t + 8 < 24) ? (t + 8) : t;
    const _Float16* wn_next = wt + (size_t)((tn >> 1) * 16) * KP;

    v8f acc = {};
#pragma unroll
    for (int ks = 0; ks < 6; ++ks) {
      v16h bnext;
      if (ks < 5) bnext = load_frag_h(wn + (ks + 1) * 32, KP, lane);
      else        bnext = load_frag_h(wn_next, KP, lane);
      acc = __builtin_amdgcn_wmma_f32_16x16x32_f16(false, afrag[ks], false, bcur, (short)0, acc, false, false);
      bcur = bnext;
    }

    int col = n0 + (lane & 15);
    if (col < DIM) {
      float bias = proj_b[col];
#pragma unroll
      for (int r = 0; r < 8; ++r) {
        int row = m0 + msub * 16 + r + 8 * (lane >> 4);
        out[(size_t)row * DIM + col] = acc[r] + bias;
      }
    }
  }
}

// ---------------------------------------------------------------------------
extern "C" void kernel_launch(void* const* d_in, const int* in_sizes, int n_in,
                              void* d_out, int out_size, void* d_ws, size_t ws_size,
                              hipStream_t stream) {
  const float* x          = (const float*)d_in[0];
  const float* qkv_w      = (const float*)d_in[1];
  const float* qkv_b      = (const float*)d_in[2];
  const float* proj_w     = (const float*)d_in[3];
  const float* proj_b     = (const float*)d_in[4];
  const float* bias_table = (const float*)d_in[5];
  const int*   rel_index  = (const int*)d_in[6];
  const float* mask       = (const float*)d_in[7];
  float* out = (float*)d_out;

  // workspace layout (all offsets 256-aligned)
  char* ws = (char*)d_ws;
  _Float16* qkv_wt  = (_Float16*)(ws);                       // 576*192 f16 (padded rows => OOB-safe frag loads)
  _Float16* proj_wt = (_Float16*)(ws + 221184);              // 192*192 f16
  float*    rel_b   = (float*)   (ws + 221184 + 73728);      // 6*64*64 f32
  size_t off = 393216;
  const size_t qkvsz = (size_t)NWIN * HEADS * NTOK * HDP * sizeof(_Float16);  // 100.66 MB each
  _Float16* qb  = (_Float16*)(ws + off); off += qkvsz;
  _Float16* kb  = (_Float16*)(ws + off); off += qkvsz;
  _Float16* vtb = (_Float16*)(ws + off); off += qkvsz;
  _Float16* ao  = (_Float16*)(ws + off);                     // 262144*192 f16

  prep_kernel<<<(NQKV * KP + 255) / 256, 256, 0, stream>>>(
      qkv_w, proj_w, bias_table, rel_index, qkv_wt, proj_wt, rel_b);

  qkv_gemm_kernel<<<MROWS / 32, 256, 0, stream>>>(x, qkv_wt, qkv_b, qb, kb, vtb);

  attn_kernel<<<NWIN, 256, 0, stream>>>(qb, kb, vtb, rel_b, mask, ao);

  proj_gemm_kernel<<<MROWS / 32, 256, 0, stream>>>(ao, proj_wt, proj_b, out);
}
